// GroupedExperts_17368847745263
// MI455X (gfx1250) — compile-verified, matrix operand order checked by reading, not run on hardware
//
#include <hip/hip_runtime.h>
#include <hip/hip_bf16.h>

// MoE grouped-expert FFN for gfx1250 (MI455X), bf16x2 split-precision WMMA.

typedef __attribute__((ext_vector_type(16))) __bf16 v16bf;
typedef __attribute__((ext_vector_type(8)))  float  v8f;

#define N_TOK  4096
#define TOPK   2
#define NE     8
#define HID    1024
#define INTER  2048
#define NPAIR  (N_TOK * TOPK)   // 8192 routed rows
#define BM     64
#define BN     128
#define KB     32

struct Hdr {
  int counts[NE];
  int cursors[NE];
  int offsets[NE + 1];
  int nTiles1;
  int nTiles2;
  int tileOff1[NE + 1];
  int tileOff2[NE + 1];
};

#define PERM_TOK_OFF 256
#define PERM_W_OFF   (256 + NPAIR * 4)
#define HBUF_OFF     131072              // hbuf: NPAIR x INTER f32 = 64MB

__device__ __forceinline__ unsigned short f2bf(float x) {
  unsigned int b = __float_as_uint(x);
  unsigned int r = (b + 0x7FFFu + ((b >> 16) & 1u)) >> 16;   // RNE
  return (unsigned short)r;
}
__device__ __forceinline__ float bf2f(unsigned short h) {
  return __uint_as_float(((unsigned int)h) << 16);
}

// ---------------------------------------------------------------- routing
__global__ void moe_init(Hdr* h) {
  int t = threadIdx.x;
  if (t < NE) { h->counts[t] = 0; h->cursors[t] = 0; }
}

__global__ void moe_zero_out(float* out) {
  int i = blockIdx.x * blockDim.x + threadIdx.x;
  if (i < (N_TOK * HID) / 4) {
    float4 z; z.x = 0.f; z.y = 0.f; z.z = 0.f; z.w = 0.f;
    ((float4*)out)[i] = z;
  }
}

__global__ void moe_count(const int* __restrict__ eidx, Hdr* h) {
  int t = blockIdx.x * blockDim.x + threadIdx.x;
  if (t < NPAIR) atomicAdd(&h->counts[eidx[t]], 1);
}

__global__ void moe_scan(Hdr* h) {
  if (threadIdx.x == 0) {
    int off = 0, t1 = 0, t2 = 0;
    for (int e = 0; e < NE; ++e) {
      h->offsets[e] = off; h->cursors[e] = off;
      h->tileOff1[e] = t1; h->tileOff2[e] = t2;
      int c  = h->counts[e];
      int rt = (c + BM - 1) / BM;
      off += c;
      t1 += rt * (INTER / BN);
      t2 += rt * (HID / BN);
    }
    h->offsets[NE] = off; h->tileOff1[NE] = t1; h->tileOff2[NE] = t2;
    h->nTiles1 = t1; h->nTiles2 = t2;
  }
}

__global__ void moe_scatter(const int* __restrict__ eidx, const float* __restrict__ ew,
                            Hdr* h, int* permTok, float* permW) {
  int t = blockIdx.x * blockDim.x + threadIdx.x;
  if (t < NPAIR) {
    int e = eidx[t];
    int pos = atomicAdd(&h->cursors[e], 1);
    permTok[pos] = t >> 1;        // token id (t = token*TOPK + slot)
    permW[pos]   = ew[t];
  }
}

// ---------------------------------------------------------------- GEMM1: h = silu(x@G) * (x@U)
__global__ __launch_bounds__(256) void moe_ffn1(
    const float* __restrict__ tokens, const float* __restrict__ gateW,
    const float* __restrict__ upW, const Hdr* __restrict__ hdr,
    const int* __restrict__ permTok, float* __restrict__ hbuf)
{
  __shared__ alignas(32) unsigned short aHi[4][32][16];
  __shared__ alignas(32) unsigned short aLo[4][32][16];
  __shared__ alignas(32) unsigned short gHi[8][32][16];
  __shared__ alignas(32) unsigned short gLo[8][32][16];
  __shared__ alignas(32) unsigned short uHi[8][32][16];
  __shared__ alignas(32) unsigned short uLo[8][32][16];
  __shared__ int rowTok[BM];

  const int tid  = threadIdx.x;
  const int lane = tid & 31;
  const int wv   = tid >> 5;     // 0..7
  const int rg   = wv >> 1;      // 16-row group 0..3
  const int cp   = wv & 1;       // 64-col half

  // LDS fill coordinates (fragment layout: lane = (m|n)&15 + 16*((k>>3)&1), elem = (k&7)+8*(k>>4))
  const int aRow  = tid >> 2;            // 0..63
  const int aKq   = (tid & 3) * 8;       // 0,8,16,24
  const int aLane = (aRow & 15) + (((aKq >> 3) & 1) << 4);
  const int aEl0  = (aKq >> 4) * 8;
  const int aRg   = aRow >> 4;
  const int bN    = tid & 127;           // 0..127
  const int bKq0  = (tid >> 7) * 8;      // 0 or 8
  const int bNg   = bN >> 4;
  const int bLn0  = bN & 15;

  const int nTiles = hdr->nTiles1;
  for (int t = blockIdx.x; t < nTiles; t += gridDim.x) {
    int e = 0;
    while (e < NE - 1 && t >= hdr->tileOff1[e + 1]) ++e;
    const int lt       = t - hdr->tileOff1[e];
    const int rowTile  = lt / (INTER / BN);
    const int colTile  = lt % (INTER / BN);
    const int rowStart = hdr->offsets[e] + rowTile * BM;
    const int rowEnd   = hdr->offsets[e + 1];
    const int nStart   = colTile * BN;
    const float* G = gateW + (size_t)e * HID * INTER;
    const float* U = upW   + (size_t)e * HID * INTER;

    __syncthreads();
    if (tid < BM) {
      int r = rowStart + tid;
      rowTok[tid] = (r < rowEnd) ? permTok[r] : -1;
    }

    v8f accG[4] = {};
    v8f accU[4] = {};

    for (int k0 = 0; k0 < HID; k0 += KB) {
      __syncthreads();
      { // A tile: gathered token rows, split f32 -> bf16 hi/lo
        float x[8];
        int tok = rowTok[aRow];
        if (tok >= 0) {
          const float4* p = (const float4*)(tokens + (size_t)tok * HID + k0 + aKq);
          float4 v0 = p[0], v1 = p[1];
          x[0]=v0.x; x[1]=v0.y; x[2]=v0.z; x[3]=v0.w;
          x[4]=v1.x; x[5]=v1.y; x[6]=v1.z; x[7]=v1.w;
        } else {
          #pragma unroll
          for (int j = 0; j < 8; ++j) x[j] = 0.f;
        }
        union { unsigned short s[8]; uint4 v; } hi, lo;
        #pragma unroll
        for (int j = 0; j < 8; ++j) {
          unsigned short h = f2bf(x[j]);
          hi.s[j] = h;
          lo.s[j] = f2bf(x[j] - bf2f(h));
        }
        *(uint4*)&aHi[aRg][aLane][aEl0] = hi.v;
        *(uint4*)&aLo[aRg][aLane][aEl0] = lo.v;
      }
      // B tiles (gate & up): 32 x 128, per-thread fixed n, 8 consecutive k
      #pragma unroll
      for (int half = 0; half < 2; ++half) {
        const int kq    = bKq0 + half * 16;
        const int bLane = bLn0 + (((kq >> 3) & 1) << 4);
        const int el0   = (kq >> 4) * 8;
        union { unsigned short s[8]; uint4 v; } hi, lo;
        #pragma unroll
        for (int j = 0; j < 8; ++j) {
          float v = G[(size_t)(k0 + kq + j) * INTER + nStart + bN];
          unsigned short h = f2bf(v);
          hi.s[j] = h; lo.s[j] = f2bf(v - bf2f(h));
        }
        *(uint4*)&gHi[bNg][bLane][el0] = hi.v;
        *(uint4*)&gLo[bNg][bLane][el0] = lo.v;
        #pragma unroll
        for (int j = 0; j < 8; ++j) {
          float v = U[(size_t)(k0 + kq + j) * INTER + nStart + bN];
          unsigned short h = f2bf(v);
          hi.s[j] = h; lo.s[j] = f2bf(v - bf2f(h));
        }
        *(uint4*)&uHi[bNg][bLane][el0] = hi.v;
        *(uint4*)&uLo[bNg][bLane][el0] = lo.v;
      }
      __syncthreads();
      // WMMA: 3-term split product accumulated in f32
      v16bf ah = *(const v16bf*)&aHi[rg][lane][0];
      v16bf al = *(const v16bf*)&aLo[rg][lane][0];
      #pragma unroll
      for (int c = 0; c < 4; ++c) {
        const int ng = cp * 4 + c;
        v16bf bh = *(const v16bf*)&gHi[ng][lane][0];
        v16bf bl = *(const v16bf*)&gLo[ng][lane][0];
        accG[c] = __builtin_amdgcn_wmma_f32_16x16x32_bf16(false, ah, false, bh, (short)0, accG[c], false, false);
        accG[c] = __builtin_amdgcn_wmma_f32_16x16x32_bf16(false, ah, false, bl, (short)0, accG[c], false, false);
        accG[c] = __builtin_amdgcn_wmma_f32_16x16x32_bf16(false, al, false, bh, (short)0, accG[c], false, false);
        v16bf ch = *(const v16bf*)&uHi[ng][lane][0];
        v16bf cl = *(const v16bf*)&uLo[ng][lane][0];
        accU[c] = __builtin_amdgcn_wmma_f32_16x16x32_bf16(false, ah, false, ch, (short)0, accU[c], false, false);
        accU[c] = __builtin_amdgcn_wmma_f32_16x16x32_bf16(false, ah, false, cl, (short)0, accU[c], false, false);
        accU[c] = __builtin_amdgcn_wmma_f32_16x16x32_bf16(false, al, false, ch, (short)0, accU[c], false, false);
      }
    }
    // epilogue: h = silu(g) * u  (C/D layout: M = vgpr + 8*(lane>>4), N = lane&15)
    const int mBase = rowStart + rg * 16 + ((lane >> 4) << 3);
    #pragma unroll
    for (int c = 0; c < 4; ++c) {
      const int n = nStart + (cp * 4 + c) * 16 + (lane & 15);
      #pragma unroll
      for (int j = 0; j < 8; ++j) {
        int r = mBase + j;
        if (r < rowEnd) {
          float g = accG[c][j];
          float h = (g / (1.0f + __expf(-g))) * accU[c][j];
          hbuf[(size_t)r * INTER + n] = h;
        }
      }
    }
  }
}

// ---------------------------------------------------------------- GEMM2: out[token] += w * (h @ D)
__global__ __launch_bounds__(256) void moe_ffn2(
    const float* __restrict__ hbuf, const float* __restrict__ downW,
    const Hdr* __restrict__ hdr, const int* __restrict__ permTok,
    const float* __restrict__ permW, float* __restrict__ out)
{
  __shared__ alignas(32) unsigned short aHi[4][32][16];
  __shared__ alignas(32) unsigned short aLo[4][32][16];
  __shared__ alignas(32) unsigned short bHi[8][32][16];
  __shared__ alignas(32) unsigned short bLo[8][32][16];
  __shared__ int   rowTok[BM];
  __shared__ float rowW[BM];

  const int tid  = threadIdx.x;
  const int lane = tid & 31;
  const int wv   = tid >> 5;
  const int rg   = wv >> 1;
  const int cp   = wv & 1;

  const int aRow  = tid >> 2;
  const int aKq   = (tid & 3) * 8;
  const int aLane = (aRow & 15) + (((aKq >> 3) & 1) << 4);
  const int aEl0  = (aKq >> 4) * 8;
  const int aRg   = aRow >> 4;
  const int bN    = tid & 127;
  const int bKq0  = (tid >> 7) * 8;
  const int bNg   = bN >> 4;
  const int bLn0  = bN & 15;

  const int nTiles = hdr->nTiles2;
  for (int t = blockIdx.x; t < nTiles; t += gridDim.x) {
    int e = 0;
    while (e < NE - 1 && t >= hdr->tileOff2[e + 1]) ++e;
    const int lt       = t - hdr->tileOff2[e];
    const int rowTile  = lt / (HID / BN);
    const int colTile  = lt % (HID / BN);
    const int rowStart = hdr->offsets[e] + rowTile * BM;
    const int rowEnd   = hdr->offsets[e + 1];
    const int nStart   = colTile * BN;
    const float* D = downW + (size_t)e * INTER * HID;

    __syncthreads();           // protects rowTok/rowW still read by prior epilogue
    if (tid < BM) {
      int r = rowStart + tid;
      int ok = (r < rowEnd);
      rowTok[tid] = ok ? permTok[r] : -1;
      rowW[tid]   = ok ? permW[r]   : 0.f;
    }

    v8f acc[4] = {};

    for (int k0 = 0; k0 < INTER; k0 += KB) {
      __syncthreads();
      { // A tile: contiguous hbuf rows, split hi/lo
        float x[8];
        int r = rowStart + aRow;
        if (r < rowEnd) {
          const float4* p = (const float4*)(hbuf + (size_t)r * INTER + k0 + aKq);
          float4 v0 = p[0], v1 = p[1];
          x[0]=v0.x; x[1]=v0.y; x[2]=v0.z; x[3]=v0.w;
          x[4]=v1.x; x[5]=v1.y; x[6]=v1.z; x[7]=v1.w;
        } else {
          #pragma unroll
          for (int j = 0; j < 8; ++j) x[j] = 0.f;
        }
        union { unsigned short s[8]; uint4 v; } hi, lo;
        #pragma unroll
        for (int j = 0; j < 8; ++j) {
          unsigned short h = f2bf(x[j]);
          hi.s[j] = h;
          lo.s[j] = f2bf(x[j] - bf2f(h));
        }
        *(uint4*)&aHi[aRg][aLane][aEl0] = hi.v;
        *(uint4*)&aLo[aRg][aLane][aEl0] = lo.v;
      }
      #pragma unroll
      for (int half = 0; half < 2; ++half) {
        const int kq    = bKq0 + half * 16;
        const int bLane = bLn0 + (((kq >> 3) & 1) << 4);
        const int el0   = (kq >> 4) * 8;
        union { unsigned short s[8]; uint4 v; } hi, lo;
        #pragma unroll
        for (int j = 0; j < 8; ++j) {
          float v = D[(size_t)(k0 + kq + j) * HID + nStart + bN];
          unsigned short h = f2bf(v);
          hi.s[j] = h; lo.s[j] = f2bf(v - bf2f(h));
        }
        *(uint4*)&bHi[bNg][bLane][el0] = hi.v;
        *(uint4*)&bLo[bNg][bLane][el0] = lo.v;
      }
      __syncthreads();
      v16bf ah = *(const v16bf*)&aHi[rg][lane][0];
      v16bf al = *(const v16bf*)&aLo[rg][lane][0];
      #pragma unroll
      for (int c = 0; c < 4; ++c) {
        const int ng = cp * 4 + c;
        v16bf bh = *(const v16bf*)&bHi[ng][lane][0];
        v16bf bl = *(const v16bf*)&bLo[ng][lane][0];
        acc[c] = __builtin_amdgcn_wmma_f32_16x16x32_bf16(false, ah, false, bh, (short)0, acc[c], false, false);
        acc[c] = __builtin_amdgcn_wmma_f32_16x16x32_bf16(false, ah, false, bl, (short)0, acc[c], false, false);
        acc[c] = __builtin_amdgcn_wmma_f32_16x16x32_bf16(false, al, false, bh, (short)0, acc[c], false, false);
      }
    }
    // epilogue: weighted atomic scatter into out
    const int mLoc = rg * 16 + ((lane >> 4) << 3);
    #pragma unroll
    for (int c = 0; c < 4; ++c) {
      const int n = nStart + (cp * 4 + c) * 16 + (lane & 15);
      #pragma unroll
      for (int j = 0; j < 8; ++j) {
        int rl  = mLoc + j;
        int tok = rowTok[rl];
        if (tok >= 0) {
          atomicAdd(&out[(size_t)tok * HID + n], rowW[rl] * acc[c][j]);
        }
      }
    }
  }
}

// ---------------------------------------------------------------- launch
extern "C" void kernel_launch(void* const* d_in, const int* in_sizes, int n_in,
                              void* d_out, int out_size, void* d_ws, size_t ws_size,
                              hipStream_t stream) {
  const float* tokens = (const float*)d_in[0];
  const int*   eidx   = (const int*)d_in[1];
  const float* ew     = (const float*)d_in[2];
  const float* gateW  = (const float*)d_in[3];
  const float* upW    = (const float*)d_in[4];
  const float* downW  = (const float*)d_in[5];
  float* out = (float*)d_out;

  char* ws = (char*)d_ws;
  Hdr*  hdr     = (Hdr*)ws;
  int*  permTok = (int*)(ws + PERM_TOK_OFF);
  float* permW  = (float*)(ws + PERM_W_OFF);
  float* hbuf   = (float*)(ws + HBUF_OFF);   // needs ~64.1 MB of workspace

  moe_init<<<1, 32, 0, stream>>>(hdr);
  moe_zero_out<<<(N_TOK * HID / 4 + 255) / 256, 256, 0, stream>>>(out);
  moe_count<<<NPAIR / 256, 256, 0, stream>>>(eidx, hdr);
  moe_scan<<<1, 1, 0, stream>>>(hdr);
  moe_scatter<<<NPAIR / 256, 256, 0, stream>>>(eidx, ew, hdr, permTok, permW);
  moe_ffn1<<<1024, 256, 0, stream>>>(tokens, gateW, upW, hdr, permTok, hbuf);
  moe_ffn2<<<1024, 256, 0, stream>>>(hbuf, downW, hdr, permTok, permW, out);
}
